// NSLoss_52269751992393
// MI455X (gfx1250) — matrix operand order repaired and checked by reading, not compiled, not run on hardware
//
#include <hip/hip_runtime.h>
#include <hip/hip_bf16.h>

typedef __attribute__((ext_vector_type(16))) _Float16 v16h;
typedef __attribute__((ext_vector_type(8)))  _Float16 h8;
typedef __attribute__((ext_vector_type(8)))  float    v8f;

#define N_SAMPLES 65536
#define DIN 4
#define HID 256
#define DOUT 5
#define M_TILE 16
#define THREADS 256            /* 8 waves; each wave owns 2 N-tiles */
#define NBLOCKS (N_SAMPLES / M_TILE) /* 4096 */
#define MU_CONST 1.7894e-05f

// Hardware v_tanh_f32 if the toolchain exposes it (gfx1250 TRANS op),
// else ocml tanhf. Guarded so both host and device passes compile.
#if defined(__HIP_DEVICE_COMPILE__) && __has_builtin(__builtin_amdgcn_tanhf)
#define TANH_F32(x) __builtin_amdgcn_tanhf(x)
#else
#define TANH_F32(x) tanhf(x)
#endif

// ---------------------------------------------------------------------------
// Prep: W1,W2,W3 (f32 [k][n]) -> f16 transposed [li][n][k] (WMMA B layout),
// plus W4 (f32 [k][5]) -> f16 [n(16, zero-padded)][k] for the output layer.
// ---------------------------------------------------------------------------
__global__ void prep_weights_f16t(const float* __restrict__ W1,
                                  const float* __restrict__ W2,
                                  const float* __restrict__ W3,
                                  const float* __restrict__ W4,
                                  _Float16* __restrict__ wt,
                                  _Float16* __restrict__ w4t) {
  int idx = blockIdx.x * blockDim.x + threadIdx.x;
  if (idx < 3 * HID * HID) {
    int li = idx >> 16;
    int r  = idx & 65535;
    int k  = r >> 8;      // input index
    int n  = r & 255;     // output index (coalesced on read)
    const float* W = (li == 0) ? W1 : (li == 1) ? W2 : W3;
    wt[(size_t)li * (HID * HID) + (size_t)n * HID + k] = (_Float16)W[k * HID + n];
  } else {
    int j = idx - 3 * HID * HID;      // 0 .. 16*256-1
    if (j < 16 * HID) {
      int n = j >> 8;                 // padded output column 0..15
      int k = j & 255;
      w4t[n * HID + k] = (n < DOUT) ? (_Float16)W4[k * DOUT + n] : (_Float16)0.0f;
    }
  }
}

// B fragment (32x16, KxN): element e -> K = 32c + 16*kh + e, column fixed per lane.
static __device__ __forceinline__ v16h load_bfrag(const _Float16* p) {
  const h8* bp = (const h8*)p;
  h8 lo = bp[0], hi = bp[1];
  v16h b;
#pragma unroll
  for (int i = 0; i < 8; ++i) { b[i] = lo[i]; b[8 + i] = hi[i]; }
  return b;
}

// A fragment (16x32, MxK): M = lane&15, K = 32c + 16*kh + e
static __device__ __forceinline__ v16h load_afrag(const _Float16* row, int c, int kh) {
  const h8* ap = (const h8*)row;
  h8 lo = ap[4 * c + 2 * kh];
  h8 hi = ap[4 * c + 2 * kh + 1];
  v16h a;
#pragma unroll
  for (int i = 0; i < 8; ++i) { a[i] = lo[i]; a[8 + i] = hi[i]; }
  return a;
}

// ---------------------------------------------------------------------------
// Fused PINN kernel: forward + 3 JVP tangents + 2 second-order streams,
// 6 streams x 16 samples per workgroup; hidden AND output layers via WMMA.
// Each wave owns 2 N-tiles so every A fragment is reused by 2 WMMAs.
// ---------------------------------------------------------------------------
__global__ __launch_bounds__(THREADS)
void pinn_fused(const float* __restrict__ x, const float* __restrict__ target,
                const float* __restrict__ W0, const float* __restrict__ b0,
                const float* __restrict__ b1, const float* __restrict__ b2,
                const float* __restrict__ b3, const float* __restrict__ b4,
                const _Float16* __restrict__ wt,
                const _Float16* __restrict__ w4t,
                float2* __restrict__ partials) {
  __shared__ __align__(16) _Float16 act[6][M_TILE][HID]; // 48 KB, 6 streams f16
  __shared__ float outbuf[M_TILE][6][DOUT];
  __shared__ float red[M_TILE][2];

  const int tid  = threadIdx.x;
  const int lane = tid & 31;
  const int wave = tid >> 5;       // 0..7; owns N-tiles 2*wave, 2*wave+1
  const int kh   = lane >> 4;      // K-half select (0/1)
  const int lm   = lane & 15;
  const int m0   = blockIdx.x * M_TILE;

  // ---- Layer 0 (DIN=4 -> HID): analytic tangents, no matmul needed ----
  for (int idx = tid; idx < M_TILE * HID; idx += THREADS) {
    int m = idx >> 8;
    int n = idx & 255;
    const float* xm = x + (size_t)(m0 + m) * DIN;
    float z = b0[n];
#pragma unroll
    for (int i = 0; i < DIN; ++i) z += xm[i] * W0[i * HID + n];
    float t0 = W0[0 * HID + n];
    float t1 = W0[1 * HID + n];
    float t2 = W0[2 * HID + n];
    float h  = TANH_F32(z);
    float d1 = 1.0f - h * h;
    float te = t0 + t1 + t2;
    float c2 = -2.0f * h * d1;
    act[0][m][n] = (_Float16)h;
    act[1][m][n] = (_Float16)(d1 * t0);
    act[2][m][n] = (_Float16)(d1 * t1);
    act[3][m][n] = (_Float16)(d1 * t2);
    act[4][m][n] = (_Float16)(c2 * t0 * te);   // sz=0 at first layer
    act[5][m][n] = (_Float16)(c2 * t1 * te);
  }
  __syncthreads();

  // ---- Hidden layers 1..3 (HID -> HID) via v_wmma_f32_16x16x32_f16 ----
  const int nc0 = (2 * wave + 0) * 16 + lm;  // column for N-tile 0
  const int nc1 = (2 * wave + 1) * 16 + lm;  // column for N-tile 1
  for (int li = 0; li < 3; ++li) {
    const _Float16* wl0 = wt + (size_t)li * (HID * HID) + (size_t)nc0 * HID;
    const _Float16* wl1 = wt + (size_t)li * (HID * HID) + (size_t)nc1 * HID;
    v8f acc0[6] = {};
    v8f acc1[6] = {};
    for (int c = 0; c < 8; ++c) {      // K chunks of 32
      v16h bf0 = load_bfrag(wl0 + 32 * c + 16 * kh);
      v16h bf1 = load_bfrag(wl1 + 32 * c + 16 * kh);
#pragma unroll
      for (int s = 0; s < 6; ++s) {
        v16h af = load_afrag(&act[s][lm][0], c, kh);  // loaded once, used twice
        acc0[s] = __builtin_amdgcn_wmma_f32_16x16x32_f16(
            false, af, false, bf0, (short)0, acc0[s], false, false);
        acc1[s] = __builtin_amdgcn_wmma_f32_16x16x32_f16(
            false, af, false, bf1, (short)0, acc1[s], false, false);
      }
    }
    const float* bl = (li == 0) ? b1 : (li == 1) ? b2 : b3;
    float bias0 = bl[nc0];
    float bias1 = bl[nc1];
    __syncthreads();   // all waves done reading act for this layer
    // C/D layout: VGPR r holds M = r + 8*kh, column = nc0/nc1
#pragma unroll
    for (int t = 0; t < 2; ++t) {
      const v8f* acc = t ? acc1 : acc0;
      const int nc = t ? nc1 : nc0;
      const float bias = t ? bias1 : bias0;
#pragma unroll
      for (int r = 0; r < 8; ++r) {
        int m = r + 8 * kh;
        float z   = acc[0][r] + bias;
        float tz0 = acc[1][r], tz1 = acc[2][r], tz2 = acc[3][r];
        float sz0 = acc[4][r], sz1 = acc[5][r];
        float h  = TANH_F32(z);
        float d1 = 1.0f - h * h;
        float te = tz0 + tz1 + tz2;
        float c2 = -2.0f * h * d1;
        act[0][m][nc] = (_Float16)h;
        act[1][m][nc] = (_Float16)(d1 * tz0);
        act[2][m][nc] = (_Float16)(d1 * tz1);
        act[3][m][nc] = (_Float16)(d1 * tz2);
        act[4][m][nc] = (_Float16)(d1 * sz0 + c2 * tz0 * te);
        act[5][m][nc] = (_Float16)(d1 * sz1 + c2 * tz1 * te);
      }
    }
    __syncthreads();
  }

  // ---- Output layer (HID -> 5, padded to 16) via WMMA: wave s = stream s ----
  if (wave < 6) {
    const int s = wave;
    v8f accf = {};
    for (int c = 0; c < 8; ++c) {
      v16h bf = load_bfrag(w4t + (size_t)lm * HID + 32 * c + 16 * kh);
      v16h af = load_afrag(&act[s][lm][0], c, kh);
      accf = __builtin_amdgcn_wmma_f32_16x16x32_f16(
          false, af, false, bf, (short)0, accf, false, false);
    }
    if (lm < DOUT) {
      float bias = (s == 0) ? b4[lm] : 0.0f;
#pragma unroll
      for (int r = 0; r < 8; ++r)
        outbuf[r + 8 * kh][s][lm] = accf[r] + bias;
    }
  }
  __syncthreads();

  // ---- Per-sample NS residuals + data loss ----
  if (tid < M_TILE) {
    int m = tid;
    const float* o0 = outbuf[m][0];
    float u = o0[0], v = o0[1], rho = o0[3];
    float ug0 = outbuf[m][1][0], ug1 = outbuf[m][2][0], ug2 = outbuf[m][3][0];
    float vg0 = outbuf[m][1][1], vg1 = outbuf[m][2][1], vg2 = outbuf[m][3][1];
    float wg0 = outbuf[m][1][2], wg1 = outbuf[m][2][2], wg2 = outbuf[m][3][2];
    float pg0 = outbuf[m][1][4], pg1 = outbuf[m][2][4];
    float hu = outbuf[m][4][0] + outbuf[m][5][0];   // (H_u e)_0 + (H_u e)_1
    float hv = outbuf[m][4][1] + outbuf[m][5][1];
    float hw = outbuf[m][4][2] + outbuf[m][5][2];
    float mx = rho * (ug2 + u * ug0 + v * ug1) + pg0 - MU_CONST * hu;
    float my = rho * (vg2 + u * vg0 + v * vg1) + pg1 - MU_CONST * hv;
    float mz = rho * (wg2 + u * wg0 + v * wg1) - MU_CONST * hw;
    const float* tg = target + (size_t)(m0 + m) * 3;
    float d0 = o0[0] - tg[0], d1v = o0[1] - tg[1], d2v = o0[2] - tg[2];
    red[m][0] = d0 * d0 + d1v * d1v + d2v * d2v;
    red[m][1] = mx * mx + my * my + mz * mz;
  }
  __syncthreads();
  if (tid == 0) {
    float ds = 0.0f, ps = 0.0f;
    for (int m = 0; m < M_TILE; ++m) { ds += red[m][0]; ps += red[m][1]; }
    partials[blockIdx.x] = make_float2(ds, ps);
  }
}

// ---------------------------------------------------------------------------
// Deterministic final reduction: fixed strided sums + fixed tree.
// ---------------------------------------------------------------------------
__global__ void reduce_partials(const float2* __restrict__ partials,
                                float* __restrict__ out) {
  __shared__ float sd[256], sp[256];
  int t = threadIdx.x;
  float ds = 0.0f, ps = 0.0f;
  for (int i = t; i < NBLOCKS; i += 256) { ds += partials[i].x; ps += partials[i].y; }
  sd[t] = ds; sp[t] = ps;
  __syncthreads();
  for (int off = 128; off > 0; off >>= 1) {
    if (t < off) { sd[t] += sd[t + off]; sp[t] += sp[t + off]; }
    __syncthreads();
  }
  if (t == 0) {
    float data = sd[0] / (3.0f * (float)N_SAMPLES);
    float phys = sp[0] / (float)N_SAMPLES;
    out[0] = data + phys;
    out[1] = data;
    out[2] = phys;
  }
}

extern "C" void kernel_launch(void* const* d_in, const int* in_sizes, int n_in,
                              void* d_out, int out_size, void* d_ws, size_t ws_size,
                              hipStream_t stream) {
  (void)in_sizes; (void)n_in; (void)out_size; (void)ws_size;
  const float* x      = (const float*)d_in[0];
  const float* target = (const float*)d_in[1];
  const float* W0 = (const float*)d_in[2];
  const float* b0 = (const float*)d_in[3];
  const float* W1 = (const float*)d_in[4];
  const float* b1 = (const float*)d_in[5];
  const float* W2 = (const float*)d_in[6];
  const float* b2 = (const float*)d_in[7];
  const float* W3 = (const float*)d_in[8];
  const float* b3 = (const float*)d_in[9];
  const float* W4 = (const float*)d_in[10];
  const float* b4 = (const float*)d_in[11];

  // d_ws: [0,32KB) partials; [32KB,+384KB) f16 W1..W3 [n][k]; then 8KB W4 [16][k]
  float2*   partials = (float2*)d_ws;
  _Float16* wt  = (_Float16*)((char*)d_ws + (size_t)NBLOCKS * sizeof(float2));
  _Float16* w4t = wt + (size_t)3 * HID * HID;

  int prep_elems = 3 * HID * HID + 16 * HID;
  prep_weights_f16t<<<(prep_elems + 255) / 256, 256, 0, stream>>>(
      W1, W2, W3, W4, wt, w4t);
  pinn_fused<<<NBLOCKS, THREADS, 0, stream>>>(x, target, W0, b0, b1, b2, b3, b4,
                                              wt, w4t, partials);
  reduce_partials<<<1, 256, 0, stream>>>(partials, (float*)d_out);
}
